// multimodes_critic_70420283785767
// MI455X (gfx1250) — compile-verified, matrix operand order checked by reading, not run on hardware
//
#include <hip/hip_runtime.h>
#include <hip/hip_bf16.h>

// ---------------------------------------------------------------------------
// Problem constants
// ---------------------------------------------------------------------------
#define BB   8
#define NN   2048
#define NNP  1024
#define FN   6
#define FS   3
#define FP   6
#define HH   32
#define NQ   128
#define C1   288   // 9 fused layer-1 branches * 32
#define C2   256   // 8 fused layer-2 A_n branches * 32
#define CTOT 352   // 11 * 32 pooled features

// LDS geometry for the GEMM kernel: rows padded to 80 B (conflict-free b128)
#define ROWB    80
#define SMB_OFF (128 * ROWB)           // B tile base within one buffer
#define BUFB    (SMB_OFF + 32 * ROWB)  // one double-buffer slice = 12800 B

// ---------------------------------------------------------------------------
// Types for WMMA
// ---------------------------------------------------------------------------
typedef float  v8f   __attribute__((ext_vector_type(8)));
typedef __bf16 v8bf  __attribute__((ext_vector_type(8)));
typedef __bf16 v16bf __attribute__((ext_vector_type(16)));

static __device__ __forceinline__ unsigned short f2bf(float f) {
    unsigned u = __float_as_uint(f);
    u += 0x7FFFu + ((u >> 16) & 1u);   // round-to-nearest-even
    return (unsigned short)(u >> 16);
}
static __device__ __forceinline__ float bf2f(unsigned short h) {
    return __uint_as_float(((unsigned)h) << 16);
}
// pack two f32 -> two bf16 (round-half-up) in ONE v_perm_b32 (+2 v_add)
static __device__ __forceinline__ unsigned pk2bf(float lo, float hi) {
    unsigned ul = __float_as_uint(lo) + 0x8000u;
    unsigned uh = __float_as_uint(hi) + 0x8000u;
    return __builtin_amdgcn_perm(uh, ul, 0x07060302u);  // {uh[31:16], ul[31:16]}
}

// ---------------------------------------------------------------------------
// Prep: fused bias vectors, colsum of W_l2[10], pool column maps, zero pools
// ---------------------------------------------------------------------------
__global__ void prep_kernel(const float* __restrict__ b_l1a, const float* __restrict__ b_l1b,
                            const float* __restrict__ b_l2,  const float* __restrict__ W_l2,
                            float* bias1, float* biasp, float* bias2an,
                            float* bias2ts, float* bias2cs, float* bias2s,
                            float* colsum10,
                            int* map_an2, int* map_ts, int* map_cs, int* map_s, int* map_p,
                            float* pooled2, float* pooled14) {
    int t = threadIdx.x;  // 1024 threads
    if (t < C1) bias1[t] = (t < 96) ? b_l1a[t] : b_l1b[t - 96];
    if (t < 32) biasp[t] = b_l1a[96 + t];
    if (t < C2) {
        int sub = t >> 5;
        int br  = (sub == 0) ? 0 : ((sub <= 6) ? (sub + 3) : 10);
        bias2an[t] = b_l2[br * 32 + (t & 31)];
    }
    if (t < 32) {
        bias2ts[t] = b_l2[32 + t];
        bias2cs[t] = b_l2[64 + t];
        bias2s[t]  = b_l2[96 + t];
        float s = 0.f;
        for (int k = 0; k < 32; ++k) s += W_l2[10 * 1024 + k * 32 + t];
        colsum10[t] = s;
    }
    if (t < 8) {
        int br = (t == 0) ? 0 : ((t <= 6) ? (t + 3) : 10);
        map_an2[t] = br * 32;
    }
    if (t == 0) { map_ts[0] = 32; map_cs[0] = 64; map_s[0] = 96; map_p[0] = 0; }
    for (int i = t; i < BB * CTOT; i += 1024) pooled2[i] = 0.f;
    if (t < BB * 32) pooled14[t] = 0.f;
}

// ---------------------------------------------------------------------------
// Layer-1 per-node projections -> P1 col-major (B, 288, N) bf16 (branch
// order: x11,x12,x13, xb0..xb5).  node is fastest -> coalesced stores, and
// the weight column is wave-uniform -> scalar loads.
// ---------------------------------------------------------------------------
__global__ void proj1_kernel(const float* __restrict__ x_n,
                             const float* __restrict__ sg, const float* __restrict__ st,
                             const float* __restrict__ og1, const float* __restrict__ ot1,
                             const float* __restrict__ og2, const float* __restrict__ ot2,
                             const float* __restrict__ W_l1a, const float* __restrict__ W_l1b,
                             unsigned short* __restrict__ P1) {
    long g = (long)blockIdx.x * 256 + threadIdx.x;
    if (g >= (long)BB * C1 * NN) return;
    int  node = (int)(g % NN);
    long rest = g / NN;
    int  col  = (int)(rest % C1);
    int  b    = (int)(rest / C1);
    int  br = col >> 5, j = col & 31;
    float s = 0.f;
    if (br < 3) {
        const float* x = x_n + ((size_t)b * NN + node) * FN;
        const float* W = W_l1a + br * (FN * HH);
        #pragma unroll
        for (int k = 0; k < FN; ++k) s += x[k] * W[k * HH + j];
    } else {
        const float* fp = (br == 3) ? sg : (br == 4) ? st : (br == 5) ? og1
                        : (br == 6) ? ot1 : (br == 7) ? og2 : ot2;
        const float* x = fp + ((size_t)b * NN + node) * FS;
        const float* W = W_l1b + (br - 3) * (FS * HH);
        #pragma unroll
        for (int k = 0; k < FS; ++k) s += x[k] * W[k * HH + j];
    }
    P1[g] = f2bf(s);
}

// second-graph projection: x_p @ W_l1a[3] -> Pp col-major (B, 32, NP) bf16
__global__ void projp_kernel(const float* __restrict__ x_p,
                             const float* __restrict__ W_l1a,
                             unsigned short* __restrict__ Pp) {
    long g = (long)blockIdx.x * 256 + threadIdx.x;
    if (g >= (long)BB * HH * NNP) return;
    int  node = (int)(g % NNP);
    long rest = g / NNP;
    int  j    = (int)(rest % HH);
    int  b    = (int)(rest / HH);
    const float* x = x_p + ((size_t)b * NNP + node) * FP;
    const float* W = W_l1a + 3 * (FP * HH);
    float s = 0.f;
    #pragma unroll
    for (int k = 0; k < FP; ++k) s += x[k] * W[k * HH + j];
    Pp[g] = f2bf(s);
}

// ---------------------------------------------------------------------------
// Core batched GEMM: Z = A(f32, n x n) @ P(bf16, col-major (B, Cp, n)) via
// v_wmma_f32_16x16x32_bf16, f32 accumulation.
//   MODE 0: X(col-major B,Cx,n) = bf16(relu(Z + bias))          (layer 1)
//   MODE 1: pooled[b, map[by]+c] += sum_rows relu(Z + bias)     (layer 2/A_p)
// Block = 256 threads (8 waves); tile = 128 rows x 32 cols; K step 32.
// Software-pipelined with DOUBLE-BUFFERED LDS (2 x 12.8 KB) -> ONE barrier
// per K-step:
//  * B tile (next step): 4x global_load_async_to_lds_b128 issued by wave 0
//    right after the barrier; s_wait_asynccnt 0 only at the end of the
//    interval, so the DMA is hidden under the WMMAs + A staging.
//  * A tile (next step): VMEM float4 loads (issue under the WMMAs),
//    f32->bf16 via v_perm packing, ds_store_b64.
//  * LDS rows padded to 80 B so b128 fragment fetches are bank-conflict free.
// ---------------------------------------------------------------------------
template <int MODE>
__global__ __launch_bounds__(256) void gcn_gemm_kernel(
    const float* __restrict__ A, int n,
    const unsigned short* __restrict__ P, int Cp,
    const float* __restrict__ bias,
    unsigned short* __restrict__ X, int Cx,
    float* __restrict__ pooled, int poolLd, const int* __restrict__ poolMap) {

    __shared__ __align__(16) unsigned char smem[2 * BUFB];

    const int t    = threadIdx.x;
    const int lane = t & 31;
    const int wv   = t >> 5;
    const int fr   = lane & 15;     // row-in-wave (A) / column (B,D)
    const int kh   = lane >> 4;     // K-half selector per documented layout
    const int b    = blockIdx.z;
    const int m0   = blockIdx.x * 128;
    const int c0   = blockIdx.y * 32;

    const float* Ab = A + (size_t)b * n * n;
    const unsigned long long Pb64 =
        (unsigned long long)(P + (size_t)b * Cp * n);      // uniform -> SGPR pair
    const unsigned voffB0 = (unsigned)((c0 + lane) * n) * 2u;

    // --- pipeline helpers -------------------------------------------------
    auto issueB = [&](int kkS, int bufW) {   // async DMA: B tile -> LDS
        if (wv == 0) {
            unsigned ldsB = (unsigned)(bufW + SMB_OFF + lane * ROWB);
            unsigned voff = voffB0 + (unsigned)kkS * 2u;
            asm volatile(
                "global_load_async_to_lds_b128 %0, %1, %2 offset:0\n\t"
                "global_load_async_to_lds_b128 %0, %1, %2 offset:16\n\t"
                "global_load_async_to_lds_b128 %0, %1, %2 offset:32\n\t"
                "global_load_async_to_lds_b128 %0, %1, %2 offset:48"
                :: "v"(ldsB), "v"(voff), "s"(Pb64) : "memory");
        }
    };
    auto waitB = [&]() {
        if (wv == 0) asm volatile("s_wait_asynccnt 0x0" ::: "memory");
    };
    auto stageA = [&](int kkS, int bufW) {   // 128x32 f32 -> bf16 in LDS
        #pragma unroll
        for (int i = 0; i < 4; ++i) {
            int lin = t + i * 256;          // 0..1023 float4 chunks
            int r = lin >> 3, kq = lin & 7;
            const float4 f = *(const float4*)(Ab + (size_t)(m0 + r) * n + kkS + kq * 4);
            unsigned p0 = pk2bf(f.x, f.y);
            unsigned p1 = pk2bf(f.z, f.w);
            *(uint2*)(smem + bufW + r * ROWB + kq * 8) = make_uint2(p0, p1);
        }
    };

    v8f acc0 = {};
    v8f acc1 = {};

    // --- prologue: fill buffer 0 for K-step 0 ---
    issueB(0, 0);
    stageA(0, 0);
    waitB();

    const int nIter = n >> 5;
    for (int it = 0; it < nIter; ++it) {
        __syncthreads();                    // buffer (it&1) ready for all waves
        const int bufR = (it & 1) * BUFB;
        const int bufW = BUFB - bufR;
        const bool more = (it + 1 < nIter);

        if (more) issueB((it + 1) << 5, bufW);   // next B tile in flight

        // --- fragment fetch per 16-bit A-matrix layout (K pairs, half-split) ---
        const unsigned char* pa = smem + bufR + (wv * 16 + fr) * ROWB + kh * 16;
        v8bf a_lo = *(const v8bf*)pa;
        v8bf a_hi = *(const v8bf*)(pa + 32);
        v16bf afrag = __builtin_shufflevector(a_lo, a_hi,
            0, 1, 2, 3, 4, 5, 6, 7, 8, 9, 10, 11, 12, 13, 14, 15);

        const unsigned char* pb0 = smem + bufR + SMB_OFF + fr * ROWB + kh * 16;
        v8bf b0_lo = *(const v8bf*)pb0;
        v8bf b0_hi = *(const v8bf*)(pb0 + 32);
        v16bf bfrag0 = __builtin_shufflevector(b0_lo, b0_hi,
            0, 1, 2, 3, 4, 5, 6, 7, 8, 9, 10, 11, 12, 13, 14, 15);

        const unsigned char* pb1 = smem + bufR + SMB_OFF + (16 + fr) * ROWB + kh * 16;
        v8bf b1_lo = *(const v8bf*)pb1;
        v8bf b1_hi = *(const v8bf*)(pb1 + 32);
        v16bf bfrag1 = __builtin_shufflevector(b1_lo, b1_hi,
            0, 1, 2, 3, 4, 5, 6, 7, 8, 9, 10, 11, 12, 13, 14, 15);

        acc0 = __builtin_amdgcn_wmma_f32_16x16x32_bf16(false, afrag, false, bfrag0,
                                                       (short)0, acc0, false, false);
        acc1 = __builtin_amdgcn_wmma_f32_16x16x32_bf16(false, afrag, false, bfrag1,
                                                       (short)0, acc1, false, false);

        if (more) {
            stageA((it + 1) << 5, bufW);    // global loads issue under the WMMAs
            waitB();                        // next-step DMA done before barrier
        }
    }

    const float bia0 = bias[c0 + fr];
    const float bia1 = bias[c0 + 16 + fr];

    if (MODE == 0) {
        // D layout: lane -> column (fr), VGPR v -> row v + 8*kh. Col-major X
        // makes each lane's 8 rows contiguous: 8 bf16 = one b128 store.
        const int mrow = m0 + wv * 16 + kh * 8;
        uint4 pk0, pk1;
        pk0.x = pk2bf(fmaxf(acc0[0] + bia0, 0.f), fmaxf(acc0[1] + bia0, 0.f));
        pk0.y = pk2bf(fmaxf(acc0[2] + bia0, 0.f), fmaxf(acc0[3] + bia0, 0.f));
        pk0.z = pk2bf(fmaxf(acc0[4] + bia0, 0.f), fmaxf(acc0[5] + bia0, 0.f));
        pk0.w = pk2bf(fmaxf(acc0[6] + bia0, 0.f), fmaxf(acc0[7] + bia0, 0.f));
        pk1.x = pk2bf(fmaxf(acc1[0] + bia1, 0.f), fmaxf(acc1[1] + bia1, 0.f));
        pk1.y = pk2bf(fmaxf(acc1[2] + bia1, 0.f), fmaxf(acc1[3] + bia1, 0.f));
        pk1.z = pk2bf(fmaxf(acc1[4] + bia1, 0.f), fmaxf(acc1[5] + bia1, 0.f));
        pk1.w = pk2bf(fmaxf(acc1[6] + bia1, 0.f), fmaxf(acc1[7] + bia1, 0.f));
        *(uint4*)(X + ((size_t)b * Cx + c0 + fr)      * n + mrow) = pk0;
        *(uint4*)(X + ((size_t)b * Cx + c0 + 16 + fr) * n + mrow) = pk1;
    } else {
        float s0 = 0.f, s1 = 0.f;
        #pragma unroll
        for (int v = 0; v < 8; ++v) {
            s0 += fmaxf(acc0[v] + bia0, 0.f);
            s1 += fmaxf(acc1[v] + bia1, 0.f);
        }
        s0 += __shfl_xor(s0, 16, 32);   // combine row halves (m 0-7 with 8-15)
        s1 += __shfl_xor(s1, 16, 32);
        if (kh == 0) {
            int base = poolMap[blockIdx.y];
            atomicAdd(&pooled[(size_t)b * poolLd + base + fr],      s0);
            atomicAdd(&pooled[(size_t)b * poolLd + base + 16 + fr], s1);
        }
    }
}

// ---------------------------------------------------------------------------
// Layer-2 per-node projections from fused layer-1 activations X1 (bf16,
// col-major (B, 288, N)).  Outputs col-major Q buffers.
// Q_an (B,256,N): [x11*W2_0, xb0..5*W2_4..9, x14*W2_10]; x14 row is constant
//   pooled14[b, n>>6] -> projection = scalar * colsum(W_l2[10]).
// Q_ts/cs/s (B,32,N): x12*W2_1, x12*W2_2, x13*W2_3.
// ---------------------------------------------------------------------------
__global__ void proj2_kernel(const unsigned short* __restrict__ X1,
                             const float* __restrict__ pooled14,
                             const float* __restrict__ colsum10,
                             const float* __restrict__ W_l2,
                             unsigned short* __restrict__ Qan,
                             unsigned short* __restrict__ Qts,
                             unsigned short* __restrict__ Qcs,
                             unsigned short* __restrict__ Qs) {
    long g = (long)blockIdx.x * 256 + threadIdx.x;
    if (g >= (long)BB * CTOT * NN) return;
    int  node = (int)(g % NN);
    long rest = g / NN;
    int  col  = (int)(rest % CTOT);
    int  b    = (int)(rest / CTOT);

    if (col < C2) {
        int sub = col >> 5, j = col & 31;
        float s;
        if (sub == 7) {   // x14 branch: constant row * column-sum of W_l2[10]
            s = pooled14[b * 32 + (node >> 6)] * colsum10[j];
        } else {
            int srcCol = (sub == 0) ? 0 : (sub + 2) * 32;          // x11 or xb_{sub-1}
            int wIdx   = (sub == 0) ? 0 : (sub + 3);               // W_l2[0], W_l2[4..9]
            const unsigned short* x = X1 + ((size_t)b * C1 + srcCol) * NN + node;
            const float* W = W_l2 + wIdx * (HH * HH);
            s = 0.f;
            #pragma unroll
            for (int k = 0; k < HH; ++k) s += bf2f(x[(size_t)k * NN]) * W[k * HH + j];
        }
        Qan[((size_t)b * C2 + col) * NN + node] = f2bf(s);
    } else {
        int c2 = col - C2;
        int grp = c2 >> 5, j = c2 & 31;                            // 0:ts 1:cs 2:s
        int srcCol = (grp < 2) ? 32 : 64;                          // x12, x12, x13
        const unsigned short* x = X1 + ((size_t)b * C1 + srcCol) * NN + node;
        const float* W = W_l2 + (grp + 1) * (HH * HH);             // W_l2[1..3]
        float s = 0.f;
        #pragma unroll
        for (int k = 0; k < HH; ++k) s += bf2f(x[(size_t)k * NN]) * W[k * HH + j];
        unsigned short* Q = (grp == 0) ? Qts : (grp == 1) ? Qcs : Qs;
        Q[((size_t)b * HH + j) * NN + node] = f2bf(s);
    }
}

// ---------------------------------------------------------------------------
// MLP head: q(8,352) -> relu(qWd1+bd1) -> relu(.Wd2+bd2) -> .Wo+bo -> (8,1)
// ---------------------------------------------------------------------------
__global__ void head_kernel(const float* __restrict__ pooled2,
                            const float* __restrict__ Wd1, const float* __restrict__ bd1,
                            const float* __restrict__ Wd2, const float* __restrict__ bd2,
                            const float* __restrict__ Wo,  const float* __restrict__ bo,
                            float* __restrict__ out) {
    __shared__ float q1[BB][NQ];
    __shared__ float red[BB][NQ];
    int t = threadIdx.x;  // 128
    for (int b = 0; b < BB; ++b) {
        float s = bd1[t];
        for (int k = 0; k < CTOT; ++k) s += pooled2[b * CTOT + k] * Wd1[k * NQ + t];
        q1[b][t] = fmaxf(s, 0.f);
    }
    __syncthreads();
    for (int b = 0; b < BB; ++b) {
        float s = bd2[t];
        for (int k = 0; k < NQ; ++k) s += q1[b][k] * Wd2[k * NQ + t];
        red[b][t] = fmaxf(s, 0.f) * Wo[t];
    }
    __syncthreads();
    if (t < BB) {
        float s = bo[0];
        for (int k = 0; k < NQ; ++k) s += red[t][k];
        out[t] = s;
    }
}

// ---------------------------------------------------------------------------
// Launch
// ---------------------------------------------------------------------------
extern "C" void kernel_launch(void* const* d_in, const int* in_sizes, int n_in,
                              void* d_out, int out_size, void* d_ws, size_t ws_size,
                              hipStream_t stream) {
    const float* x_n    = (const float*)d_in[0];
    const float* A_n    = (const float*)d_in[1];
    const float* A_s    = (const float*)d_in[2];
    const float* A_n_ts = (const float*)d_in[3];
    const float* A_n_cs = (const float*)d_in[4];
    /* mask d_in[5] unused by the model */
    const float* x_p    = (const float*)d_in[6];
    const float* A_p    = (const float*)d_in[7];
    const float* self_g   = (const float*)d_in[8];
    const float* self_t   = (const float*)d_in[9];
    const float* other_g1 = (const float*)d_in[10];
    const float* other_t1 = (const float*)d_in[11];
    const float* other_g2 = (const float*)d_in[12];
    const float* other_t2 = (const float*)d_in[13];
    const float* W_l1a = (const float*)d_in[14];
    const float* b_l1a = (const float*)d_in[15];
    const float* W_l1b = (const float*)d_in[16];
    const float* b_l1b = (const float*)d_in[17];
    const float* W_l2  = (const float*)d_in[18];
    const float* b_l2  = (const float*)d_in[19];
    const float* Wd1   = (const float*)d_in[20];
    const float* bd1   = (const float*)d_in[21];
    const float* Wd2   = (const float*)d_in[22];
    const float* bd2   = (const float*)d_in[23];
    const float* Wo    = (const float*)d_in[24];
    const float* bo    = (const float*)d_in[25];
    float* out = (float*)d_out;

    unsigned char* ws = (unsigned char*)d_ws;
    size_t cur = 0;
    auto alloc = [&](size_t bytes) -> void* {
        void* p = ws + cur;
        cur = (cur + bytes + 255) & ~(size_t)255;
        return p;
    };

    unsigned short* P1  = (unsigned short*)alloc((size_t)BB * NN * C1 * 2);
    unsigned short* Pp  = (unsigned short*)alloc((size_t)BB * NNP * HH * 2);
    unsigned short* X1  = (unsigned short*)alloc((size_t)BB * NN * C1 * 2);
    unsigned short* Qan = (unsigned short*)alloc((size_t)BB * NN * C2 * 2);
    unsigned short* Qts = (unsigned short*)alloc((size_t)BB * NN * HH * 2);
    unsigned short* Qcs = (unsigned short*)alloc((size_t)BB * NN * HH * 2);
    unsigned short* Qs  = (unsigned short*)alloc((size_t)BB * NN * HH * 2);
    float* pooled2  = (float*)alloc(BB * CTOT * 4);
    float* pooled14 = (float*)alloc(BB * 32 * 4);
    float* bias1    = (float*)alloc(C1 * 4);
    float* biasp    = (float*)alloc(32 * 4);
    float* bias2an  = (float*)alloc(C2 * 4);
    float* bias2ts  = (float*)alloc(32 * 4);
    float* bias2cs  = (float*)alloc(32 * 4);
    float* bias2s   = (float*)alloc(32 * 4);
    float* colsum10 = (float*)alloc(32 * 4);
    int* map_an2 = (int*)alloc(8 * 4);
    int* map_ts  = (int*)alloc(4);
    int* map_cs  = (int*)alloc(4);
    int* map_s   = (int*)alloc(4);
    int* map_p   = (int*)alloc(4);

    // 1. prep (fused biases, colsums, pool maps, zero pool accumulators)
    prep_kernel<<<1, 1024, 0, stream>>>(b_l1a, b_l1b, b_l2, W_l2,
        bias1, biasp, bias2an, bias2ts, bias2cs, bias2s, colsum10,
        map_an2, map_ts, map_cs, map_s, map_p, pooled2, pooled14);

    // 2. layer-1 per-node projections (bf16, col-major)
    {
        long tot = (long)BB * C1 * NN;
        proj1_kernel<<<dim3((unsigned)((tot + 255) / 256)), 256, 0, stream>>>(
            x_n, self_g, self_t, other_g1, other_t1, other_g2, other_t2,
            W_l1a, W_l1b, P1);
        long totp = (long)BB * HH * NNP;
        projp_kernel<<<dim3((unsigned)((totp + 255) / 256)), 256, 0, stream>>>(
            x_p, W_l1a, Pp);
    }

    // 3. fused layer-1 GCN on main graph: X1 = relu(A_n @ P1 + bias1)   [WMMA]
    gcn_gemm_kernel<0><<<dim3(NN / 128, C1 / 32, BB), 256, 0, stream>>>(
        A_n, NN, P1, C1, bias1, X1, C1, nullptr, 0, nullptr);

    // 4. second-graph GCN + global sum pool -> pooled14                 [WMMA]
    gcn_gemm_kernel<1><<<dim3(NNP / 128, 1, BB), 256, 0, stream>>>(
        A_p, NNP, Pp, HH, biasp, nullptr, 0, pooled14, 32, map_p);

    // 5. layer-2 per-node projections (bf16, col-major)
    {
        long tot = (long)BB * CTOT * NN;
        proj2_kernel<<<dim3((unsigned)((tot + 255) / 256)), 256, 0, stream>>>(
            X1, pooled14, colsum10, W_l2, Qan, Qts, Qcs, Qs);
    }

    // 6. layer-2 GCNs + node pooling straight into pooled2              [WMMA]
    gcn_gemm_kernel<1><<<dim3(NN / 128, C2 / 32, BB), 256, 0, stream>>>(
        A_n, NN, Qan, C2, bias2an, nullptr, 0, pooled2, CTOT, map_an2);
    gcn_gemm_kernel<1><<<dim3(NN / 128, 1, BB), 256, 0, stream>>>(
        A_n_ts, NN, Qts, HH, bias2ts, nullptr, 0, pooled2, CTOT, map_ts);
    gcn_gemm_kernel<1><<<dim3(NN / 128, 1, BB), 256, 0, stream>>>(
        A_n_cs, NN, Qcs, HH, bias2cs, nullptr, 0, pooled2, CTOT, map_cs);
    gcn_gemm_kernel<1><<<dim3(NN / 128, 1, BB), 256, 0, stream>>>(
        A_s, NN, Qs, HH, bias2s, nullptr, 0, pooled2, CTOT, map_s);

    // 7. MLP head -> out (8,1)
    head_kernel<<<1, NQ, 0, stream>>>(pooled2, Wd1, bd1, Wd2, bd2, Wo, bo, out);
}
